// EmbeddingLayer_35948876268299
// MI455X (gfx1250) — matrix operand order repaired and checked by reading, not compiled, not run on hardware
//
#include <hip/hip_runtime.h>
#include <hip/hip_bf16.h>

// ---------------------------------------------------------------------------
// Problem constants (from the reference)
// ---------------------------------------------------------------------------
#define BB 8
#define LL 128
#define SS 160
#define HH 768
#define TT 16
#define CD 50
#define WD 100
#define NSEQ (BB * LL)          // 1024 char sequences
#define GOUT 200                // 4*CD gate outputs
#define GPAD 208                // padded to 13 tiles of 16
#define KPAD 64                 // CD=50 padded to 64 (2 x K=32)
#define EW (HH + WD + 2 * CD)   // 968 embed width
#define NEGV (-1e30f)

typedef __attribute__((ext_vector_type(16))) _Float16 v16h;
typedef __attribute__((ext_vector_type(8)))  float    v8f;

// Load 16 halfs from two 16-byte-aligned chunks into a v16h (8 VGPRs).
__device__ __forceinline__ v16h load2x8h(const _Float16* p0, const _Float16* p1) {
    v16h r;
    reinterpret_cast<float4*>(&r)[0] = *reinterpret_cast<const float4*>(p0);
    reinterpret_cast<float4*>(&r)[1] = *reinterpret_cast<const float4*>(p1);
    return r;
}

__device__ __forceinline__ float sigmoidf(float x) {
    return 1.0f / (1.0f + expf(-x));
}

// ---------------------------------------------------------------------------
// 1) Global min of bert_emb (two stages)
// ---------------------------------------------------------------------------
__global__ void min_part_kernel(const float* __restrict__ x, int n, float* __restrict__ out) {
    __shared__ float s[256];
    float v = 3.4e38f;
    for (int i = blockIdx.x * blockDim.x + threadIdx.x; i < n; i += gridDim.x * blockDim.x)
        v = fminf(v, x[i]);
    s[threadIdx.x] = v;
    __syncthreads();
    for (int k = 128; k > 0; k >>= 1) {
        if ((int)threadIdx.x < k) s[threadIdx.x] = fminf(s[threadIdx.x], s[threadIdx.x + k]);
        __syncthreads();
    }
    if (threadIdx.x == 0) out[blockIdx.x] = s[0];
}

__global__ void min_final_kernel(const float* __restrict__ part, int n, float* __restrict__ out) {
    __shared__ float s[256];
    float v = 3.4e38f;
    for (int i = threadIdx.x; i < n; i += 256) v = fminf(v, part[i]);
    s[threadIdx.x] = v;
    __syncthreads();
    for (int k = 128; k > 0; k >>= 1) {
        if ((int)threadIdx.x < k) s[threadIdx.x] = fminf(s[threadIdx.x], s[threadIdx.x + k]);
        __syncthreads();
    }
    if (threadIdx.x == 0) out[0] = s[0];
}

// ---------------------------------------------------------------------------
// 2) Pack the four LSTM weight matrices (200x50 f32) into (208x64) f16, [N][K]
//    order: w_ih_f, w_hh_f, w_ih_b, w_hh_b
// ---------------------------------------------------------------------------
__global__ void pack_w_kernel(const float* __restrict__ wihf, const float* __restrict__ whhf,
                              const float* __restrict__ wihb, const float* __restrict__ whhb,
                              _Float16* __restrict__ wpack) {
    int idx = blockIdx.x * 256 + threadIdx.x;
    if (idx >= 4 * GPAD * KPAD) return;
    int k = idx & (KPAD - 1);
    int n = (idx >> 6) % GPAD;
    int w = idx / (GPAD * KPAD);
    const float* src = (w == 0) ? wihf : (w == 1) ? whhf : (w == 2) ? wihb : whhb;
    float v = (n < GOUT && k < CD) ? src[n * CD + k] : 0.0f;
    wpack[idx] = (_Float16)v;
}

// ---------------------------------------------------------------------------
// 3) Gather char embeddings into f16 x[N][T][64] and reversed x_rev[N][T][64]
// ---------------------------------------------------------------------------
__global__ void char_gather_kernel(const int* __restrict__ char_ids,
                                   const int* __restrict__ char_count,
                                   const float* __restrict__ char_table,
                                   _Float16* __restrict__ x, _Float16* __restrict__ xrev) {
    int idx = blockIdx.x * 256 + threadIdx.x;
    if (idx >= NSEQ * TT * KPAD) return;
    int c = idx & (KPAD - 1);
    int t = (idx >> 6) & (TT - 1);
    int n = idx >> 10;
    float v = 0.0f, vr = 0.0f;
    if (c < CD) {
        int id = char_ids[n * TT + t];
        v = char_table[id * CD + c];
        int len = max(char_count[n], 1);
        int rev = len - 1 - t;
        if (rev >= 0) {
            int idr = char_ids[n * TT + rev];
            vr = char_table[idr * CD + c];
        }
    }
    x[idx]    = (_Float16)v;
    xrev[idx] = (_Float16)vr;
}

// ---------------------------------------------------------------------------
// 4) BiLSTM via WMMA. One wave (32 threads) owns 16 sequences; blockIdx.y is
//    direction. Gates g = x_t@Wih^T + h@Whh^T + b computed as 13 N-tiles of
//    16x16, each with 4x v_wmma_f32_16x16x32_f16 (2 K-steps for x, 2 for h).
// ---------------------------------------------------------------------------
__global__ __launch_bounds__(32) void char_lstm_wmma_kernel(
    const _Float16* __restrict__ xfwd, const _Float16* __restrict__ xrev,
    const _Float16* __restrict__ wpack,
    const float* __restrict__ b_f, const float* __restrict__ b_b,
    float* __restrict__ hfwd, float* __restrict__ hrev) {

    const int lane = threadIdx.x;
    const int dir  = blockIdx.y;
    const int seqbase = blockIdx.x * 16;

    const _Float16* xbuf = dir ? xrev : xfwd;
    const _Float16* Wx   = wpack + (size_t)dir * 2 * GPAD * KPAD;
    const _Float16* Wh   = Wx + GPAD * KPAD;
    const float*    bias = dir ? b_b : b_f;
    float*          hout = dir ? hrev : hfwd;

    __shared__ __align__(16) _Float16 h_lds[16][KPAD];
    __shared__ float g_lds[16][GPAD];
    __shared__ float c_lds[16][CD];

    for (int i = lane; i < 16 * KPAD; i += 32) (&h_lds[0][0])[i] = (_Float16)0.0f;
    for (int i = lane; i < 16 * CD;   i += 32) (&c_lds[0][0])[i] = 0.0f;
    __syncthreads();

    const int m    = lane & 15;
    const int offA = (lane >= 16) ? 8 : 0;    // unused; A tiles also load with offB
    const int offB = (lane >= 16) ? 16 : 0;   // B-matrix K offset per lane-half
    const int hi   = (lane >= 16) ? 8 : 0;    // D-matrix M offset per lane-half

    for (int t = 0; t < TT; ++t) {
        // A tiles: activations for this timestep (K = 0..31 and 32..63)
        const _Float16* xrow = xbuf + ((size_t)(seqbase + m) * TT + t) * KPAD;
        v16h ax0 = load2x8h(xrow + 0  + offB, xrow + 8 + offB);
        v16h ax1 = load2x8h(xrow + 32 + offB, xrow + 40 + offB);
        const _Float16* hrow = &h_lds[m][0];
        v16h ah0 = load2x8h(hrow + 0  + offB, hrow + 8 + offB);
        v16h ah1 = load2x8h(hrow + 32 + offB, hrow + 40 + offB);

        for (int nt = 0; nt < 13; ++nt) {
            const int ncol = nt * 16 + m;
            const _Float16* wxrow = Wx + (size_t)ncol * KPAD;
            const _Float16* whrow = Wh + (size_t)ncol * KPAD;
            // B tiles: 16 contiguous halfs per lane-half at [N][K]
            v16h bx0 = load2x8h(wxrow + offB,      wxrow + offB + 8);
            v16h bx1 = load2x8h(wxrow + 32 + offB, wxrow + 40 + offB);
            v16h bh0 = load2x8h(whrow + offB,      whrow + offB + 8);
            v16h bh1 = load2x8h(whrow + 32 + offB, whrow + 40 + offB);

            float bval = (ncol < GOUT) ? bias[ncol] : 0.0f;
            v8f acc;
            #pragma unroll
            for (int r = 0; r < 8; ++r) acc[r] = bval;

            acc = __builtin_amdgcn_wmma_f32_16x16x32_f16(false, ax0, false, bx0, (short)0, acc, false, false);
            acc = __builtin_amdgcn_wmma_f32_16x16x32_f16(false, ax1, false, bx1, (short)0, acc, false, false);
            acc = __builtin_amdgcn_wmma_f32_16x16x32_f16(false, ah0, false, bh0, (short)0, acc, false, false);
            acc = __builtin_amdgcn_wmma_f32_16x16x32_f16(false, ah1, false, bh1, (short)0, acc, false, false);

            #pragma unroll
            for (int r = 0; r < 8; ++r) g_lds[hi + r][ncol] = acc[r];
        }
        __syncthreads();

        // LSTM pointwise update: 16 rows x 50 hidden units, 25 per lane.
        for (int idx = lane; idx < 16 * CD; idx += 32) {
            int mm = idx / CD;
            int j  = idx - mm * CD;
            float ig = g_lds[mm][j];
            float fg = g_lds[mm][CD + j];
            float gg = g_lds[mm][2 * CD + j];
            float og = g_lds[mm][3 * CD + j];
            float c  = sigmoidf(fg) * c_lds[mm][j] + sigmoidf(ig) * tanhf(gg);
            float h  = sigmoidf(og) * tanhf(c);
            c_lds[mm][j] = c;
            h_lds[mm][j] = (_Float16)h;
            hout[((size_t)(seqbase + mm) * TT + t) * CD + j] = h;
        }
        __syncthreads();
    }
}

// ---------------------------------------------------------------------------
// 5) word_reps: masked max over S pieces, one block per (b,l)
// ---------------------------------------------------------------------------
__global__ void word_reps_kernel(const float* __restrict__ bert, const int* __restrict__ p2w,
                                 const float* __restrict__ minv_p, float* __restrict__ out) {
    int bl = blockIdx.x;
    int b = bl >> 7;  // / LL
    __shared__ int smask[SS];
    for (int s = threadIdx.x; s < SS; s += 256) smask[s] = p2w[(size_t)bl * SS + s];
    __syncthreads();
    float minv = minv_p[0];
    const float* base = bert + (size_t)b * SS * HH;
    for (int h = threadIdx.x; h < HH; h += 256) {
        float acc = minv;
        for (int s = 0; s < SS; ++s)
            if (smask[s]) acc = fmaxf(acc, base[(size_t)s * HH + h]);
        out[(size_t)bl * EW + h] = acc;
    }
}

// ---------------------------------------------------------------------------
// 6) cls_embedding copy
// ---------------------------------------------------------------------------
__global__ void cls_kernel(const float* __restrict__ bert, float* __restrict__ out) {
    int idx = blockIdx.x * 256 + threadIdx.x;
    if (idx >= BB * HH) return;
    int b = idx / HH, h = idx - b * HH;
    out[(size_t)NSEQ * EW + idx] = bert[(size_t)b * SS * HH + h];
}

// ---------------------------------------------------------------------------
// 7) word_embed gather + char max-pool into output
// ---------------------------------------------------------------------------
__global__ void tail_kernel(const int* __restrict__ word_ids, const float* __restrict__ word_table,
                            const int* __restrict__ char_count, const int* __restrict__ tmask,
                            const float* __restrict__ hfwd, const float* __restrict__ hrev,
                            float* __restrict__ out) {
    int idx = blockIdx.x * 256 + threadIdx.x;
    if (idx >= NSEQ * (2 * CD)) return;
    int n = idx / (2 * CD);
    int c = idx - n * (2 * CD);

    out[(size_t)n * EW + HH + c] = word_table[(size_t)word_ids[n] * WD + c];

    int len = max(char_count[n], 1);
    float acc = NEGV;
    for (int t = 0; t < TT; ++t) {
        float hv;
        if (c < CD) {
            hv = hfwd[((size_t)n * TT + t) * CD + c];
        } else {
            int rev = len - 1 - t;
            hv = (rev >= 0) ? hrev[((size_t)n * TT + rev) * CD + (c - CD)] : 0.0f;
        }
        float v = (tmask[n * TT + t] != 0) ? hv : NEGV;
        acc = fmaxf(acc, v);
    }
    out[(size_t)n * EW + HH + WD + c] = (acc == NEGV) ? 0.0f : acc;
}

// ---------------------------------------------------------------------------
// Launch
// ---------------------------------------------------------------------------
extern "C" void kernel_launch(void* const* d_in, const int* in_sizes, int n_in,
                              void* d_out, int out_size, void* d_ws, size_t ws_size,
                              hipStream_t stream) {
    const float* bert       = (const float*)d_in[0];
    const int*   p2w        = (const int*)d_in[1];
    const int*   word_ids   = (const int*)d_in[2];
    const int*   char_ids   = (const int*)d_in[3];
    const int*   char_count = (const int*)d_in[4];
    const int*   tmask      = (const int*)d_in[5];
    const float* word_table = (const float*)d_in[6];
    const float* char_table = (const float*)d_in[7];
    const float* w_ih_f     = (const float*)d_in[8];
    const float* w_hh_f     = (const float*)d_in[9];
    const float* b_f        = (const float*)d_in[10];
    const float* w_ih_b     = (const float*)d_in[11];
    const float* w_hh_b     = (const float*)d_in[12];
    const float* b_b        = (const float*)d_in[13];
    float* out = (float*)d_out;

    // workspace carve-up
    char* ws = (char*)d_ws;
    constexpr size_t OFF_MINPART = 0;                                  // 1024 f32
    constexpr size_t OFF_MINFIN  = 4096;                               // 1 f32
    constexpr size_t OFF_WPACK   = 8192;                               // 4*208*64 f16
    constexpr size_t WPACK_BYTES = (size_t)4 * GPAD * KPAD * 2;
    constexpr size_t OFF_XFWD    = OFF_WPACK + WPACK_BYTES;            // N*T*64 f16
    constexpr size_t XBYTES      = (size_t)NSEQ * TT * KPAD * 2;
    constexpr size_t OFF_XREV    = OFF_XFWD + XBYTES;
    constexpr size_t OFF_HFWD    = OFF_XREV + XBYTES;                  // N*T*50 f32
    constexpr size_t HBYTES      = (size_t)NSEQ * TT * CD * 4;
    constexpr size_t OFF_HREV    = OFF_HFWD + HBYTES;

    float*    minpart = (float*)(ws + OFF_MINPART);
    float*    minfin  = (float*)(ws + OFF_MINFIN);
    _Float16* wpack   = (_Float16*)(ws + OFF_WPACK);
    _Float16* xfwd    = (_Float16*)(ws + OFF_XFWD);
    _Float16* xrev    = (_Float16*)(ws + OFF_XREV);
    float*    hfwd    = (float*)(ws + OFF_HFWD);
    float*    hrev    = (float*)(ws + OFF_HREV);

    const int bert_n = BB * SS * HH;

    min_part_kernel<<<1024, 256, 0, stream>>>(bert, bert_n, minpart);
    min_final_kernel<<<1, 256, 0, stream>>>(minpart, 1024, minfin);

    {
        int n = 4 * GPAD * KPAD;
        pack_w_kernel<<<(n + 255) / 256, 256, 0, stream>>>(w_ih_f, w_hh_f, w_ih_b, w_hh_b, wpack);
    }
    {
        int n = NSEQ * TT * KPAD;
        char_gather_kernel<<<(n + 255) / 256, 256, 0, stream>>>(char_ids, char_count, char_table,
                                                                xfwd, xrev);
    }

    dim3 lgrid(NSEQ / 16, 2);
    char_lstm_wmma_kernel<<<lgrid, 32, 0, stream>>>(xfwd, xrev, wpack, b_f, b_b, hfwd, hrev);

    word_reps_kernel<<<NSEQ, 256, 0, stream>>>(bert, p2w, minfin, out);
    cls_kernel<<<(BB * HH + 255) / 256, 256, 0, stream>>>(bert, out);
    {
        int n = NSEQ * 2 * CD;
        tail_kernel<<<(n + 255) / 256, 256, 0, stream>>>(word_ids, word_table, char_count, tmask,
                                                         hfwd, hrev, out);
    }
    (void)in_sizes; (void)n_in; (void)out_size; (void)ws_size;
}
